// MultiScaleSelfAttention_66589172957414
// MI455X (gfx1250) — matrix operand order repaired and checked by reading, not compiled, not run on hardware
//
#include <hip/hip_runtime.h>
#include <hip/hip_bf16.h>
#include <math.h>

typedef __attribute__((ext_vector_type(16))) _Float16 v16h;
typedef __attribute__((ext_vector_type(8)))  float    v8f;

union F16Frag {
    v16h v;
    _Float16 h[16];
    uint4 q[2];
};

// Low 32 bits of the flat shared-aperture address == LDS byte offset (ISA 10.2)
__device__ __forceinline__ uint32_t lds_addr_of(const void* p) {
    return (uint32_t)(uintptr_t)p;
}

// ---------------------------------------------------------------------------
// Fused QKV projection: for 16 rows of x, compute Q/K/V = x*W? + b? (f16 out).
// A-fragment (from f32 x) is built ONCE per k-step and feeds three WMMAs.
// Block = 128 threads = 4 waves; wave w owns output columns [16w, 16w+16).
// Grid: (M/16). K multiple of 32. KD fixed at 64.
// ---------------------------------------------------------------------------
__global__ __launch_bounds__(128) void qkv_gemm_wmma(
    const float* __restrict__ x,
    const float* __restrict__ Wq, const float* __restrict__ bq,
    const float* __restrict__ Wk, const float* __restrict__ bk,
    const float* __restrict__ Wv, const float* __restrict__ bv,
    _Float16* __restrict__ Qo, _Float16* __restrict__ Ko,
    _Float16* __restrict__ Vo, int M, int K)
{
    constexpr int KD = 64;
    const int lane = threadIdx.x & 31;
    const int wave = threadIdx.x >> 5;
    const int lm = lane & 15;
    const int lh = lane >> 4;
    const int m0 = blockIdx.x * 16;
    const int n  = wave * 16 + lm;

    const float* __restrict__ arow = x + (size_t)(m0 + lm) * K;

    v8f accq = {}, acck = {}, accv = {};
    for (int k0 = 0; k0 < K; k0 += 32) {
        if (k0 + 32 < K) __builtin_prefetch(arow + k0 + 32, 0, 0);

        // Shared A fragment: x[m][k0+16*lh..+15] (contiguous)
        F16Frag a;
        const int ka = k0 + lh * 16;
#pragma unroll
        for (int t = 0; t < 8; ++t) a.h[t]     = (_Float16)arow[ka + t];
#pragma unroll
        for (int t = 0; t < 8; ++t) a.h[8 + t] = (_Float16)arow[ka + 8 + t];

        const int kb = k0 + lh * 16;
        F16Frag b;
#pragma unroll
        for (int t = 0; t < 16; ++t) b.h[t] = (_Float16)Wq[(size_t)(kb + t) * KD + n];
        accq = __builtin_amdgcn_wmma_f32_16x16x32_f16(false, a.v, false, b.v,
                                                      (short)0, accq, false, false);
#pragma unroll
        for (int t = 0; t < 16; ++t) b.h[t] = (_Float16)Wk[(size_t)(kb + t) * KD + n];
        acck = __builtin_amdgcn_wmma_f32_16x16x32_f16(false, a.v, false, b.v,
                                                      (short)0, acck, false, false);
#pragma unroll
        for (int t = 0; t < 16; ++t) b.h[t] = (_Float16)Wv[(size_t)(kb + t) * KD + n];
        accv = __builtin_amdgcn_wmma_f32_16x16x32_f16(false, a.v, false, b.v,
                                                      (short)0, accv, false, false);
    }

    const float bnq = bq[n], bnk = bk[n], bnv = bv[n];
#pragma unroll
    for (int i = 0; i < 8; ++i) {
        const size_t row = (size_t)(m0 + lh * 8 + i);
        Qo[row * KD + n] = (_Float16)(accq[i] + bnq);
        Ko[row * KD + n] = (_Float16)(acck[i] + bnk);
        Vo[row * KD + n] = (_Float16)(accv[i] + bnv);
    }
}

// ---------------------------------------------------------------------------
// Generic GEMM: D[M,Nout] = A[M,K] * W[K,Nout] + bias (A f16, D f32 here).
// Block = 128 threads = 4 waves. Grid: (M/16, Nout/64).
// ---------------------------------------------------------------------------
template <typename AT, typename OT>
__global__ __launch_bounds__(128) void gemm_bias_wmma(
    const AT* __restrict__ A, const float* __restrict__ W,
    const float* __restrict__ bias, OT* __restrict__ D,
    int M, int K, int Nout)
{
    const int lane = threadIdx.x & 31;
    const int wave = threadIdx.x >> 5;
    const int lm = lane & 15;
    const int lh = lane >> 4;
    const int m0 = blockIdx.x * 16;
    const int n  = blockIdx.y * 64 + wave * 16 + lm;

    const AT* __restrict__ arow = A + (size_t)(m0 + lm) * K;

    v8f acc = {};
    for (int k0 = 0; k0 < K; k0 += 32) {
        F16Frag a;
        const int ka = k0 + lh * 16;
#pragma unroll
        for (int t = 0; t < 8; ++t) a.h[t]     = (_Float16)arow[ka + t];
#pragma unroll
        for (int t = 0; t < 8; ++t) a.h[8 + t] = (_Float16)arow[ka + 8 + t];

        F16Frag b;
        const int kb = k0 + lh * 16;
#pragma unroll
        for (int t = 0; t < 16; ++t)
            b.h[t] = (_Float16)W[(size_t)(kb + t) * Nout + n];

        acc = __builtin_amdgcn_wmma_f32_16x16x32_f16(
            false, a.v, false, b.v, (short)0, acc, false, false);
    }

    const float bn = bias[n];
#pragma unroll
    for (int i = 0; i < 8; ++i) {
        const int row = m0 + lh * 8 + i;
        if (row < M)
            D[(size_t)row * Nout + n] = (OT)(acc[i] + bn);
    }
}

// ---------------------------------------------------------------------------
// Flash attention (single head, KD=64), one wave per (16 query rows, batch).
// K tile staged with gfx1250 async global->LDS copies (ASYNCcnt).
// Grid: (ceil(N/16), B), block = 32.
// ---------------------------------------------------------------------------
__global__ __launch_bounds__(32) void flash_attn_wmma(
    const _Float16* __restrict__ Qh, const _Float16* __restrict__ Kh,
    const _Float16* __restrict__ Vh, _Float16* __restrict__ Oh, int N)
{
    constexpr int KD = 64;
    const int lane = threadIdx.x & 31;
    const int lm = lane & 15;
    const int lh = lane >> 4;
    const int m0 = blockIdx.x * 16;
    const size_t base = (size_t)blockIdx.y * (size_t)N * KD;

    __shared__ __align__(16) _Float16 Qs[16 * KD];   // Q tile, row-major
    __shared__ __align__(16) _Float16 Ks[32 * KD];   // K rows, row-major
    __shared__ __align__(16) _Float16 Vts[KD * 32];  // V tile transposed: [d][k]
    __shared__ __align__(16) _Float16 Ps[16 * 32];   // probabilities tile

    // ---- stage Q tile (16 rows x 64), clamped rows for the ragged tail ----
    {
        const int r = lm;
        const int dh = lh * 32;
        int gr = m0 + r; if (gr > N - 1) gr = N - 1;
        const uint4* src = (const uint4*)(Qh + base + (size_t)gr * KD + dh);
        uint4* dst = (uint4*)(Qs + r * KD + dh);
#pragma unroll
        for (int t = 0; t < 4; ++t) dst[t] = src[t];
    }
    __syncthreads();

    // ---- build both Q A-fragments (d-halves 0..31 and 32..63) ----
    v16h qa[2];
#pragma unroll
    for (int h = 0; h < 2; ++h) {
        F16Frag fa;
        const _Float16* p = Qs + lm * KD + h * 32 + lh * 16;
        fa.q[0] = *(const uint4*)(p);
        fa.q[1] = *(const uint4*)(p + 8);
        qa[h] = fa.v;
    }

    v8f acc[4];
    float runmax[8], runsum[8];
    {
        v8f z = {};
#pragma unroll
        for (int d = 0; d < 4; ++d) acc[d] = z;
#pragma unroll
        for (int i = 0; i < 8; ++i) { runmax[i] = -3.0e38f; runsum[i] = 0.0f; }
    }

    const int nkb = (N + 31) >> 5;
    for (int kb = 0; kb < nkb; ++kb) {
        const int kbase = kb << 5;
        __syncthreads();  // previous-iteration LDS consumers are done

        // ---- stage K rows via async global->LDS (one row per lane) ----
        {
            int gr = kbase + lane; if (gr > N - 1) gr = N - 1;
            const _Float16* gk = Kh + base + (size_t)gr * KD;
            const uint32_t lk = lds_addr_of(Ks + lane * KD);
#pragma unroll
            for (int t = 0; t < 8; ++t) {
                asm volatile("global_load_async_to_lds_b128 %0, %1, off"
                             :: "v"(lk + t * 16), "v"(gk + t * 8)
                             : "memory");
            }

            // ---- V rows: load wide, scatter transposed into LDS ----
            union { uint4 q[8]; _Float16 h[64]; } vb;
            const uint4* vsrc = (const uint4*)(Vh + base + (size_t)gr * KD);
#pragma unroll
            for (int t = 0; t < 8; ++t) vb.q[t] = vsrc[t];
#pragma unroll
            for (int d = 0; d < 64; ++d) Vts[d * 32 + lane] = vb.h[d];
        }
        asm volatile("s_wait_asynccnt 0x0" ::: "memory");
        __syncthreads();

        // ---- S = Q * K^T for two 16-key groups ----
        v8f s[2];
#pragma unroll
        for (int g = 0; g < 2; ++g) {
            v8f c = {};
#pragma unroll
            for (int h = 0; h < 2; ++h) {
                F16Frag fb;  // B[d][n] = K[n][32h + kh + t], contiguous 16 halfs
                const _Float16* p = Ks + (g * 16 + lm) * KD + h * 32 + lh * 16;
                fb.q[0] = *(const uint4*)(p);
                fb.q[1] = *(const uint4*)(p + 8);
                c = __builtin_amdgcn_wmma_f32_16x16x32_f16(
                    false, qa[h], false, fb.v, (short)0, c, false, false);
            }
            s[g] = c;
        }

        // ---- scale + key mask ----
        float sv[2][8];
#pragma unroll
        for (int g = 0; g < 2; ++g) {
            const int col = kbase + g * 16 + lm;
            const bool valid = (col < N);
#pragma unroll
            for (int i = 0; i < 8; ++i)
                sv[g][i] = valid ? s[g][i] * 0.125f : -3.0e38f;
        }

        // ---- row-wise block max (rows in VGPR index; cols in 16 lanes) ----
        float bm[8];
#pragma unroll
        for (int i = 0; i < 8; ++i) bm[i] = fmaxf(sv[0][i], sv[1][i]);
#pragma unroll
        for (int msk = 1; msk <= 8; msk <<= 1) {
#pragma unroll
            for (int i = 0; i < 8; ++i)
                bm[i] = fmaxf(bm[i], __shfl_xor(bm[i], msk, 32));
        }

        // ---- online softmax update ----
        float pv[2][8], bs[8];
#pragma unroll
        for (int i = 0; i < 8; ++i) {
            const float nm = fmaxf(runmax[i], bm[i]);
            const float cf = __expf(runmax[i] - nm);
            runmax[i] = nm;
            runsum[i] *= cf;
#pragma unroll
            for (int d = 0; d < 4; ++d) acc[d][i] *= cf;
            pv[0][i] = __expf(sv[0][i] - nm);
            pv[1][i] = __expf(sv[1][i] - nm);
            bs[i] = pv[0][i] + pv[1][i];
        }
#pragma unroll
        for (int msk = 1; msk <= 8; msk <<= 1) {
#pragma unroll
            for (int i = 0; i < 8; ++i)
                bs[i] += __shfl_xor(bs[i], msk, 32);
        }
#pragma unroll
        for (int i = 0; i < 8; ++i) runsum[i] += bs[i];

        // ---- P: C-layout -> LDS -> A-layout ----
#pragma unroll
        for (int g = 0; g < 2; ++g)
#pragma unroll
            for (int i = 0; i < 8; ++i)
                Ps[(lh * 8 + i) * 32 + g * 16 + lm] = (_Float16)pv[g][i];
        __syncthreads();

        F16Frag fp;
        {
            const _Float16* pp = Ps + lm * 32 + lh * 16;
            fp.q[0] = *(const uint4*)(pp);
            fp.q[1] = *(const uint4*)(pp + 8);
        }

        // ---- acc += P * V  (4 output d-slabs of 16) ----
#pragma unroll
        for (int ds = 0; ds < 4; ++ds) {
            F16Frag fb;  // B[k][n] = V[kbase+kh+t][16*ds+lm] = Vts[d][kh+t]
            const _Float16* vp = Vts + (ds * 16 + lm) * 32 + lh * 16;
            fb.q[0] = *(const uint4*)(vp);
            fb.q[1] = *(const uint4*)(vp + 8);
            acc[ds] = __builtin_amdgcn_wmma_f32_16x16x32_f16(
                false, fp.v, false, fb.v, (short)0, acc[ds], false, false);
        }
    }

    // ---- normalize and store ----
#pragma unroll
    for (int ds = 0; ds < 4; ++ds)
#pragma unroll
        for (int i = 0; i < 8; ++i) {
            const int row = m0 + lh * 8 + i;
            if (row < N)
                Oh[base + (size_t)row * KD + ds * 16 + lm] =
                    (_Float16)(acc[ds][i] / runsum[i]);
        }
}

// ---------------------------------------------------------------------------
// Host-side orchestration
// ---------------------------------------------------------------------------
extern "C" void kernel_launch(void* const* d_in, const int* in_sizes, int n_in,
                              void* d_out, int out_size, void* d_ws, size_t ws_size,
                              hipStream_t stream)
{
    (void)in_sizes; (void)n_in; (void)out_size; (void)ws_size;
    constexpr int BSZ = 64, KD = 64;
    const int Ns[3] = {784, 196, 49};
    const int Cs[3] = {256, 512, 1024};

    _Float16* ws = (_Float16*)d_ws;
    float* out = (float*)d_out;

    // workspace layout: per scale: q, k, v, attn  (each M*KD f16)
    size_t off = 0;
    size_t qOff[3], kOff[3], vOff[3], aOff[3];
    size_t oOff[3]; size_t oacc = 0;
    for (int i = 0; i < 3; ++i) {
        const size_t M = (size_t)BSZ * Ns[i];
        qOff[i] = off; off += M * KD;
        kOff[i] = off; off += M * KD;
        vOff[i] = off; off += M * KD;
        aOff[i] = off; off += M * KD;
        oOff[i] = oacc; oacc += M * Cs[i];
    }

    for (int i = 0; i < 3; ++i) {
        const float* x  = (const float*)d_in[9 * i + 0];
        const float* Wq = (const float*)d_in[9 * i + 1];
        const float* bq = (const float*)d_in[9 * i + 2];
        const float* Wk = (const float*)d_in[9 * i + 3];
        const float* bk = (const float*)d_in[9 * i + 4];
        const float* Wv = (const float*)d_in[9 * i + 5];
        const float* bv = (const float*)d_in[9 * i + 6];
        const float* Wo = (const float*)d_in[9 * i + 7];
        const float* bo = (const float*)d_in[9 * i + 8];

        const int N = Ns[i], C = Cs[i];
        const int M = BSZ * N;

        // Fused QKV projection: x read once, A-fragment reused for 3 WMMAs
        dim3 gq(M / 16), b128(128);
        qkv_gemm_wmma<<<gq, b128, 0, stream>>>(
            x, Wq, bq, Wk, bk, Wv, bv,
            ws + qOff[i], ws + kOff[i], ws + vOff[i], M, C);

        // fused softmax(QK^T/8) V
        dim3 ga((N + 15) / 16, BSZ);
        flash_attn_wmma<<<ga, 32, 0, stream>>>(
            ws + qOff[i], ws + kOff[i], ws + vOff[i], ws + aOff[i], N);

        // output projection: [M,64] x [64,C] + bo -> f32 out
        dim3 go(M / 16, C / 64);
        gemm_bias_wmma<_Float16, float><<<go, b128, 0, stream>>>(
            ws + aOff[i], Wo, bo, out + oOff[i], M, KD, C);
    }
}
